// PVAM_57836029608001
// MI455X (gfx1250) — matrix operand order repaired and checked by reading, not compiled, not run on hardware
//
#include <hip/hip_runtime.h>
#include <hip/hip_bf16.h>

// ---------------------------------------------------------------------------
// PVAM (parallel visual attention) for MI455X / gfx1250.
//   b=32, c=512, h*w=t=256, L=25
//   enc   = transpose(x) + emb            (16.8 MB)
//   feats = enc @ feat_w^T + feat_b       (8192x512x512 GEMM -> v_wmma bf16)
//   scores= sum_c tanh(tok+feats)*scw     (105M TRANS ops, VALU-bound)
//   out   = softmax_t(scores) @ enc       (fused per (b,l) block)
// ---------------------------------------------------------------------------

#define Bc  32
#define Cc  512
#define Tc  256
#define Lc  25
#define Mc  (Bc * Tc)   // 8192 GEMM rows

typedef __attribute__((ext_vector_type(8)))  float  v8f;
typedef __attribute__((ext_vector_type(16))) __bf16 v16bf;

union FragBF16 {
    uint4 q[2];
    v16bf v;
};

static __device__ __forceinline__ unsigned short f32_to_bf16_rne(float f) {
    unsigned int u = __float_as_uint(f);
    u = (u + 0x7FFFu + ((u >> 16) & 1u)) >> 16;   // round to nearest even
    return (unsigned short)u;
}

// ---------------------------------------------------------------------------
// Kernel 1: enc[b,t,c] = x[b,c,t] + emb[t,c]; emit f32 + bf16 copies.
// 32x32 LDS tile transpose so both global read (over t) and write (over c)
// are coalesced.  grid = (t/32, c/32, b), block = (32, 8).
// ---------------------------------------------------------------------------
__global__ void pvam_prep(const float* __restrict__ x,
                          const float* __restrict__ emb,
                          float* __restrict__ enc,
                          unsigned short* __restrict__ encbf) {
    __shared__ float tile[32][33];
    const int tx = threadIdx.x, ty = threadIdx.y;
    const int t0 = blockIdx.x * 32;
    const int c0 = blockIdx.y * 32;
    const int b  = blockIdx.z;

    const float* xb = x + ((size_t)b * Cc + c0) * Tc + t0;
#pragma unroll
    for (int r = 0; r < 4; ++r) {
        const int ci = ty + 8 * r;
        tile[ci][tx] = xb[(size_t)ci * Tc + tx];     // coalesced over t
    }
    __syncthreads();
#pragma unroll
    for (int r = 0; r < 4; ++r) {
        const int ti = ty + 8 * r;
        const int t = t0 + ti, c = c0 + tx;
        const float v = tile[tx][ti] + emb[(size_t)t * Cc + c];
        const size_t idx = ((size_t)b * Tc + t) * Cc + c;   // coalesced over c
        enc[idx]   = v;
        encbf[idx] = f32_to_bf16_rne(v);
    }
}

// ---------------------------------------------------------------------------
// Kernel 2: feat_w (512x512 f32, row-major [out,in]) -> bf16.
// ---------------------------------------------------------------------------
__global__ void pvam_convw(const float* __restrict__ w,
                           unsigned short* __restrict__ wbf) {
    const int i = blockIdx.x * 256 + threadIdx.x;
    wbf[i] = f32_to_bf16_rne(w[i]);
}

// ---------------------------------------------------------------------------
// Kernel 3: feats = enc @ feat_w^T + feat_b via v_wmma_f32_16x16x32_bf16.
// A (enc) is M x K row-major bf16, B (feat_w) is N x K row-major bf16 (the
// torch [out,in] layout is already K-contiguous per output row, which is
// exactly what the WMMA B fragment wants).
// One wave computes a 16x16 D tile; 8 waves/block span 128 N-columns.
// grid = (M/16, N/128), block = 256.
//
// Fragment layouts per CDNA5 ISA 7.12.2 (wave32):
//   A 16x32 bf16 : lane m=lane&15, half=lane>>4; elems 0..7  = K  8h..8h+7
//                                                elems 8..15 = K 16+8h..16+8h+7
//   B 32x16 bf16 : lane n=lane&15; same K order as A (elems 0..7 = K 8h.., 8..15 = K 16+8h..)
//   C/D 16x16 f32: VGPR r -> M = r + 8h, N = lane&15
// ---------------------------------------------------------------------------
__global__ void pvam_gemm_wmma(const unsigned short* __restrict__ A,
                               const unsigned short* __restrict__ Bw,
                               const float* __restrict__ bias,
                               float* __restrict__ C) {
    constexpr int K = Cc, N = Cc;
    const int lane = threadIdx.x & 31;
    const int wave = threadIdx.x >> 5;
    const int half = lane >> 4;
    const int row  = lane & 15;
    const int m0 = blockIdx.x * 16;
    const int n0 = blockIdx.y * 128 + wave * 16;

    const unsigned short* aPtr = A  + (size_t)(m0 + row) * K + 8  * half;
    const unsigned short* bPtr = Bw + (size_t)(n0 + row) * K + 8  * half;

    v8f acc = {0.f, 0.f, 0.f, 0.f, 0.f, 0.f, 0.f, 0.f};

#pragma unroll 4
    for (int k = 0; k < K; k += 32) {
        FragBF16 a, b;
        a.q[0] = *(const uint4*)(aPtr + k);          // K = k+8h   .. +7
        a.q[1] = *(const uint4*)(aPtr + k + 16);     // K = k+16+8h.. +7
        const uint4* bp = (const uint4*)(bPtr + k);  // K = k+8h   .. +7
        b.q[0] = bp[0];
        b.q[1] = *(const uint4*)(bPtr + k + 16);     // K = k+16+8h.. +7
        acc = __builtin_amdgcn_wmma_f32_16x16x32_bf16(
            /*neg_a=*/false, a.v, /*neg_b=*/false, b.v,
            /*c_mod=*/(short)0, acc, /*reuse_a=*/false, /*reuse_b=*/false);
    }

    const float bn = bias[n0 + row];
    float* cBase = C + (size_t)(m0 + 8 * half) * N + (n0 + row);
#pragma unroll
    for (int r = 0; r < 8; ++r)
        cBase[(size_t)r * N] = acc[r] + bn;
}

// ---------------------------------------------------------------------------
// Kernel 4: fused scores -> softmax -> weighted sum.  One block per (l, b).
// Stage 1: 8 waves each own score rows t (wave-strided); 32 lanes stride c,
//          wave-reduce via shfl_xor (wave32).
// Stage 2: block softmax over t=256 in LDS.
// Stage 3: out[b,l,c] = sum_t w[t]*enc[b,t,c], c coalesced across threads.
// ---------------------------------------------------------------------------
__global__ void pvam_attn(const float* __restrict__ feats,
                          const float* __restrict__ enc,
                          const float* __restrict__ token_w,
                          const float* __restrict__ score_w,
                          float* __restrict__ out) {
    const int l = blockIdx.x;       // 0..24
    const int b = blockIdx.y;       // 0..31
    const int tid  = threadIdx.x;   // 0..255
    const int lane = tid & 31;
    const int wave = tid >> 5;

    __shared__ float s_tok[Cc];
    __shared__ float s_scw[Cc];
    __shared__ float s_sc[Tc];
    __shared__ float s_red[Tc];
    __shared__ float s_w[Tc];

    s_tok[tid]       = token_w[(size_t)l * Cc + tid];
    s_tok[tid + 256] = token_w[(size_t)l * Cc + tid + 256];
    s_scw[tid]       = score_w[tid];
    s_scw[tid + 256] = score_w[tid + 256];
    __syncthreads();

    const float* fb = feats + (size_t)b * Tc * Cc;
    for (int t = wave; t < Tc; t += 8) {
        const float* fr = fb + (size_t)t * Cc;
        float p = 0.f;
#pragma unroll 4
        for (int c = lane; c < Cc; c += 32)
            p += tanhf(s_tok[c] + fr[c]) * s_scw[c];
        for (int off = 16; off > 0; off >>= 1)
            p += __shfl_xor(p, off, 32);
        if (lane == 0) s_sc[t] = p;
    }
    __syncthreads();

    // softmax over t
    s_red[tid] = s_sc[tid];
    __syncthreads();
    for (int s = 128; s > 0; s >>= 1) {
        if (tid < s) s_red[tid] = fmaxf(s_red[tid], s_red[tid + s]);
        __syncthreads();
    }
    const float mx = s_red[0];
    __syncthreads();
    const float e = __expf(s_sc[tid] - mx);
    s_red[tid] = e;
    __syncthreads();
    for (int s = 128; s > 0; s >>= 1) {
        if (tid < s) s_red[tid] += s_red[tid + s];
        __syncthreads();
    }
    s_w[tid] = e * (1.f / s_red[0]);
    __syncthreads();

    // weighted sum over enc
    const float* eb = enc + (size_t)b * Tc * Cc;
    for (int c = tid; c < Cc; c += 256) {
        float acc = 0.f;
#pragma unroll 4
        for (int t = 0; t < Tc; ++t)
            acc = fmaf(s_w[t], eb[(size_t)t * Cc + c], acc);
        out[((size_t)b * Lc + l) * Cc + c] = acc;
    }
}

// ---------------------------------------------------------------------------
// Workspace layout (bytes):
//   [0,            16.78 MB) enc  f32   (8192*512*4)
//   [16.78 MB,     33.55 MB) feats f32  (8192*512*4)
//   [33.55 MB,     41.94 MB) enc  bf16  (8192*512*2)
//   [41.94 MB,     42.47 MB) feat_w bf16 (512*512*2)
// ---------------------------------------------------------------------------
extern "C" void kernel_launch(void* const* d_in, const int* in_sizes, int n_in,
                              void* d_out, int out_size, void* d_ws, size_t ws_size,
                              hipStream_t stream) {
    const float* x       = (const float*)d_in[0];
    const float* emb_w   = (const float*)d_in[1];
    const float* feat_w  = (const float*)d_in[2];
    const float* feat_b  = (const float*)d_in[3];
    const float* token_w = (const float*)d_in[4];
    const float* score_w = (const float*)d_in[5];
    float* out = (float*)d_out;

    const size_t encF32Bytes = (size_t)Mc * Cc * sizeof(float);
    const size_t encBfBytes  = (size_t)Mc * Cc * sizeof(unsigned short);
    const size_t wBfBytes    = (size_t)Cc * Cc * sizeof(unsigned short);
    if (ws_size < 2 * encF32Bytes + encBfBytes + wBfBytes) return;

    char* ws = (char*)d_ws;
    float*          enc   = (float*)ws;
    float*          feats = (float*)(ws + encF32Bytes);
    unsigned short* encbf = (unsigned short*)(ws + 2 * encF32Bytes);
    unsigned short* wbf   = (unsigned short*)(ws + 2 * encF32Bytes + encBfBytes);

    pvam_prep<<<dim3(Tc / 32, Cc / 32, Bc), dim3(32, 8), 0, stream>>>(
        x, emb_w, enc, encbf);

    pvam_convw<<<(Cc * Cc) / 256, 256, 0, stream>>>(feat_w, wbf);

    pvam_gemm_wmma<<<dim3(Mc / 16, Cc / 128), 256, 0, stream>>>(
        encbf, wbf, feat_b, feats);

    pvam_attn<<<dim3(Lc, Bc), 256, 0, stream>>>(
        feats, enc, token_w, score_w, out);
}
